// SSMRec_33621003993788
// MI455X (gfx1250) — matrix operand order repaired and checked by reading, not compiled, not run on hardware
//
#include <hip/hip_runtime.h>
#include <hip/hip_bf16.h>

typedef __attribute__((ext_vector_type(16))) _Float16 v16h;
typedef __attribute__((ext_vector_type(8)))  float    v8f;

#define B_SZ  64
#define SEQ   2048
#define HDIM  256
#define NST   256
#define NBLK  2
#define NCAND 101
#define MROWS (B_SZ * SEQ)   // 131072

// Toggle: stage GEMM weights into LDS with CDNA5 async-to-LDS (ASYNCcnt path).
#define USE_ASYNC_LDS 1

union FragH { v16h v; uint4 u[2]; };

// ---------------------------------------------------------------------------
// Weight prep for block `blk` (see round-0 notes; coef broadcasts on h-axis).
// ---------------------------------------------------------------------------
__global__ void prep_weights(int blk,
    const float* __restrict__ logAr, const float* __restrict__ Aim,
    const float* __restrict__ Bssm,  const float* __restrict__ Cssm,
    const float* __restrict__ logstep,
    const float* __restrict__ W1,    const float* __restrict__ W2,
    float2* __restrict__ Abar, _Float16* __restrict__ Wb,
    _Float16* __restrict__ Wc, _Float16* __restrict__ W1t,
    _Float16* __restrict__ W2t)
{
  int tid = blockIdx.x * blockDim.x + threadIdx.x;   // 0..131071
  float step = expf(logstep[blk]);
  if (tid < NST) {
    float ar = -expf(logAr[blk * NST + tid]);
    float ai = Aim[blk * NST + tid];
    float mag = expf(ar * step);
    Abar[tid] = make_float2(mag * cosf(ai * step), mag * sinf(ai * step));
  }
  { // Wb: [512, 256]
    int row = tid >> 8;          // 0..511
    int h   = tid & 255;
    int n   = row & 255;
    float ar = -expf(logAr[blk * NST + h]);
    float ai = Aim[blk * NST + h];
    float mag = expf(ar * step);
    float abr = mag * cosf(ai * step), abi = mag * sinf(ai * step);
    float nr = abr - 1.f, ni = abi;
    float den = ar * ar + ai * ai;
    float cr = (nr * ar + ni * ai) / den;
    float ci = (ni * ar - nr * ai) / den;
    float bv = Bssm[(size_t)(blk * NST + n) * HDIM + h];
    Wb[tid] = (_Float16)((row < 256) ? cr * bv : ci * bv);
  }
  { // Wc: [256, 512]
    int h = tid >> 9;            // 0..255
    int c = tid & 511;
    int n = c & 255;
    float v = Cssm[((size_t)((blk * HDIM + h) * NST) + n) * 2 + (c >= 256 ? 1 : 0)];
    Wc[tid] = (_Float16)(c >= 256 ? -v : v);
  }
  if (tid < HDIM * HDIM) {
    int j = tid >> 8, ii = tid & 255;
    W1t[tid] = (_Float16)W1[(size_t)blk * HDIM * HDIM + ii * HDIM + j];
    W2t[tid] = (_Float16)W2[(size_t)blk * HDIM * HDIM + ii * HDIM + j];
  }
}

__global__ void embed_mask(const int* __restrict__ seqs,
                           const float* __restrict__ item_emb,
                           const float* __restrict__ pos_emb,
                           float* __restrict__ x)
{
  int tid = blockIdx.x * 256 + threadIdx.x;
  int m = tid >> 8, hh = tid & 255;
  int s = m & (SEQ - 1);
  int it = seqs[m];
  x[tid] = (it != 0) ? (item_emb[(size_t)it * HDIM + hh] + pos_emb[s * HDIM + hh])
                     : 0.f;
}

// LayerNorm, one wave32 per 256-wide row; out f16. Physical row = row*stride+off.
__global__ void ln_rows(const float* __restrict__ X, const float* __restrict__ g,
                        const float* __restrict__ bta, _Float16* __restrict__ out,
                        int rowStride, int rowOffset)
{
  int wave = threadIdx.x >> 5, lane = threadIdx.x & 31;
  int row = blockIdx.x * 8 + wave;
  const float* xp = X + ((size_t)row * rowStride + rowOffset) * HDIM + lane * 8;
  float v[8]; float s = 0.f, s2 = 0.f;
#pragma unroll
  for (int i = 0; i < 8; ++i) { v[i] = xp[i]; s += v[i]; s2 += v[i] * v[i]; }
#pragma unroll
  for (int off = 16; off; off >>= 1) {
    s  += __shfl_xor(s,  off, 32);
    s2 += __shfl_xor(s2, off, 32);
  }
  float mean = s * (1.f / HDIM);
  float inv  = rsqrtf(fmaxf(s2 * (1.f / HDIM) - mean * mean, 0.f) + 1e-5f);
  _Float16* op = out + (size_t)row * HDIM + lane * 8;
#pragma unroll
  for (int i = 0; i < 8; ++i)
    op[i] = (_Float16)((v[i] - mean) * inv * g[lane * 8 + i] + bta[lane * 8 + i]);
}

// ---------------------------------------------------------------------------
// f16 GEMM, f32 accumulate: out[m,n] = sum_k A[m,k] * Wt[n,k] (+bias, relu)
// Block = 256 threads = 8 waves; wave -> 32x64 output tile (8 wmma accums,
// each B fragment reused twice). Weights for this block's 64 columns staged
// in LDS, via async-to-LDS (ASYNCcnt) when enabled.
// grid = (M/256, Ncols/64). K in {256,512}.
// ---------------------------------------------------------------------------
__global__ __launch_bounds__(256) void gemm_f16(
    const _Float16* __restrict__ A, const _Float16* __restrict__ Wt,
    float* __restrict__ outF32, _Float16* __restrict__ outF16,
    const float* __restrict__ bias, int doRelu, int Ncols, int K)
{
  extern __shared__ _Float16 smem[];
  int wave = threadIdx.x >> 5, lane = threadIdx.x & 31;
  int n0 = blockIdx.y * 64;

  // Stage Wt[n0..n0+63][0..K-1] into LDS; 16 bytes per lane per iteration.
  int nchunks = (64 * K) >> 3;
  for (int i = threadIdx.x; i < nchunks; i += 256) {
    int f = i << 3;
    int row = f / K, col = f - row * K;
    const _Float16* src = Wt + (size_t)(n0 + row) * K + col;
#if USE_ASYNC_LDS
    unsigned ldsoff = (unsigned)(size_t)(smem + f);
    asm volatile("global_load_async_to_lds_b128 %0, %1, off"
                 :: "v"(ldsoff), "v"(src) : "memory");
#else
    ((uint4*)smem)[i] = *(const uint4*)src;
#endif
  }
#if USE_ASYNC_LDS
  asm volatile("s_wait_asynccnt 0x0" ::: "memory");
#endif
  __syncthreads();

  int m0 = blockIdx.x * 256 + wave * 32;
  v8f acc[2][4];
#pragma unroll
  for (int mt = 0; mt < 2; ++mt)
#pragma unroll
    for (int nt = 0; nt < 4; ++nt)
#pragma unroll
      for (int r = 0; r < 8; ++r) acc[mt][nt][r] = 0.f;

  int hi = lane >> 4;   // 0: lanes 0-15, 1: lanes 16-31
  // A fragment (16-bit A 16x32): lanes<16 K={k0..k0+7, k0+16..k0+23};
  // lanes>=16 shifted +8.
  const _Float16* abase0 = A + (size_t)(m0 + (lane & 15)) * K + (hi ? 8 : 0);
  const _Float16* abase1 = abase0 + (size_t)16 * K;

  for (int k0 = 0; k0 < K; k0 += 32) {
    FragH a0, a1;
    a0.u[0] = *(const uint4*)(abase0 + k0);
    a0.u[1] = *(const uint4*)(abase0 + k0 + 16);
    a1.u[0] = *(const uint4*)(abase1 + k0);
    a1.u[1] = *(const uint4*)(abase1 + k0 + 16);
    // Preload all four B fragments so ds_loads clause and WMMAs burst.
    FragH b[4];
#pragma unroll
    for (int nt = 0; nt < 4; ++nt) {
      const _Float16* bp =
          smem + (size_t)(nt * 16 + (lane & 15)) * K + k0 + (hi ? 8 : 0);
      b[nt].u[0] = *(const uint4*)bp;
      b[nt].u[1] = *(const uint4*)(bp + 16);
    }
#pragma unroll
    for (int nt = 0; nt < 4; ++nt)
      acc[0][nt] = __builtin_amdgcn_wmma_f32_16x16x32_f16(
          false, a0.v, false, b[nt].v, (short)0, acc[0][nt], false, false);
#pragma unroll
    for (int nt = 0; nt < 4; ++nt)
      acc[1][nt] = __builtin_amdgcn_wmma_f32_16x16x32_f16(
          false, a1.v, false, b[nt].v, (short)0, acc[1][nt], false, false);
  }

  // C/D layout: VGPR r -> M = base + r + 8*hi, N = n0 + nt*16 + (lane&15).
#pragma unroll
  for (int mt = 0; mt < 2; ++mt) {
    int rbase = m0 + mt * 16 + (hi ? 8 : 0);
#pragma unroll
    for (int nt = 0; nt < 4; ++nt) {
      int col = n0 + nt * 16 + (lane & 15);
      float bv = bias ? bias[col] : 0.f;
#pragma unroll
      for (int r = 0; r < 8; ++r) {
        float val = acc[mt][nt][r] + bv;
        if (doRelu) val = fmaxf(val, 0.f);
        size_t idx = (size_t)(rbase + r) * Ncols + col;
        if (outF16) outF16[idx] = (_Float16)val;
        else        outF32[idx] = val;
      }
    }
  }
}

// Diagonal complex scan: one thread per (batch, state-n) chain.
__global__ void s4_scan(const _Float16* __restrict__ Bu, _Float16* __restrict__ xs,
                        const float2* __restrict__ Abar)
{
  int n = threadIdx.x;         // 0..255
  int b = blockIdx.x;          // 0..63
  float2 ab = Abar[n];
  float xr = 0.f, xi = 0.f;
  size_t base = (size_t)b * SEQ * 512 + n;
  for (int t = 0; t < SEQ; ++t) {
    __builtin_prefetch(Bu + base + 4 * 512, 0, 0);   // global_prefetch_b8
    float br = (float)Bu[base], bi = (float)Bu[base + 256];
    float nr = fmaf(ab.x, xr, fmaf(-ab.y, xi, br));
    float ni = fmaf(ab.x, xi, fmaf( ab.y, xr, bi));
    xs[base]       = (_Float16)nr;
    xs[base + 256] = (_Float16)ni;
    xr = nr; xi = ni;
    base += 512;
  }
}

// Fused: x += y + D[h]*u, then LayerNorm(x) -> out (f16). One wave per row.
__global__ void resid_s4_ln(float* __restrict__ x, const float* __restrict__ y,
                            const _Float16* __restrict__ hln,
                            const float* __restrict__ D,
                            const float* __restrict__ g,
                            const float* __restrict__ bta,
                            _Float16* __restrict__ out, int blk)
{
  int wave = threadIdx.x >> 5, lane = threadIdx.x & 31;
  int row = blockIdx.x * 8 + wave;
  size_t base = (size_t)row * HDIM + lane * 8;
  int hb = blk * HDIM + lane * 8;
  float v[8]; float s = 0.f, s2 = 0.f;
#pragma unroll
  for (int i = 0; i < 8; ++i) {
    v[i] = x[base + i] + y[base + i] + D[hb + i] * (float)hln[base + i];
    x[base + i] = v[i];
    s += v[i]; s2 += v[i] * v[i];
  }
#pragma unroll
  for (int off = 16; off; off >>= 1) {
    s  += __shfl_xor(s,  off, 32);
    s2 += __shfl_xor(s2, off, 32);
  }
  float mean = s * (1.f / HDIM);
  float inv  = rsqrtf(fmaxf(s2 * (1.f / HDIM) - mean * mean, 0.f) + 1e-5f);
#pragma unroll
  for (int i = 0; i < 8; ++i)
    out[base + i] =
        (_Float16)((v[i] - mean) * inv * g[lane * 8 + i] + bta[lane * 8 + i]);
}

// x = mask * (x + y)   (y already has b2 from the GEMM epilogue)
__global__ void resid_ffn_mask(float* __restrict__ x, const float* __restrict__ y,
                               const int* __restrict__ seqs)
{
  int tid = blockIdx.x * 256 + threadIdx.x;
  int m = tid >> 8;
  float v = x[tid] + y[tid];
  x[tid] = (seqs[m] != 0) ? v : 0.f;
}

// logits[b,c] = dot(xl[b], item_emb[item_idxs[b,c]]); one wave per (b,c).
__global__ void logits_kernel(const _Float16* __restrict__ xl,
                              const float* __restrict__ item_emb,
                              const int* __restrict__ idxs, float* __restrict__ out)
{
  int wave = threadIdx.x >> 5, lane = threadIdx.x & 31;
  int w = blockIdx.x * 8 + wave;       // 0..6463
  int b = w / NCAND, c = w % NCAND;
  int it = idxs[b * NCAND + c];
  const _Float16* xp = xl + b * HDIM + lane * 8;
  const float*    ep = item_emb + (size_t)it * HDIM + lane * 8;
  float s = 0.f;
#pragma unroll
  for (int i = 0; i < 8; ++i) s += (float)xp[i] * ep[i];
#pragma unroll
  for (int off = 16; off; off >>= 1) s += __shfl_xor(s, off, 32);
  if (lane == 0) out[b * NCAND + c] = s;
}

extern "C" void kernel_launch(void* const* d_in, const int* in_sizes, int n_in,
                              void* d_out, int out_size, void* d_ws, size_t ws_size,
                              hipStream_t stream)
{
  (void)in_sizes; (void)n_in; (void)out_size; (void)ws_size;
  const int*   seqs     = (const int*)d_in[0];
  const int*   idxs     = (const int*)d_in[1];
  const float* item_emb = (const float*)d_in[2];
  const float* pos_emb  = (const float*)d_in[3];
  const float* ln_g     = (const float*)d_in[4];
  const float* ln_b     = (const float*)d_in[5];
  const float* logAr    = (const float*)d_in[6];
  const float* Aim      = (const float*)d_in[7];
  const float* Bssm     = (const float*)d_in[8];
  const float* Cssm     = (const float*)d_in[9];
  const float* Dssm     = (const float*)d_in[10];
  const float* logstep  = (const float*)d_in[11];
  const float* W1       = (const float*)d_in[12];
  const float* b1       = (const float*)d_in[13];
  const float* W2       = (const float*)d_in[14];
  const float* b2       = (const float*)d_in[15];
  const float* lnfg     = (const float*)d_in[16];
  const float* lnfb     = (const float*)d_in[17];

  char* ws = (char*)d_ws;
  size_t off = 0;
  auto carve = [&](size_t bytes) -> char* {
    char* p = ws + off; off += (bytes + 255) & ~(size_t)255; return p;
  };
  float2*   Abar = (float2*)  carve((size_t)NST * sizeof(float2));
  _Float16* Wb   = (_Float16*)carve((size_t)2 * NST * HDIM * 2);     // [512,256]
  _Float16* Wc   = (_Float16*)carve((size_t)HDIM * 2 * NST * 2);     // [256,512]
  _Float16* W1t  = (_Float16*)carve((size_t)HDIM * HDIM * 2);
  _Float16* W2t  = (_Float16*)carve((size_t)HDIM * HDIM * 2);
  _Float16* xl   = (_Float16*)carve((size_t)B_SZ * HDIM * 2);
  float*    x    = (float*)   carve((size_t)MROWS * HDIM * 4);       // 134 MB
  _Float16* hbuf = (_Float16*)carve((size_t)MROWS * HDIM * 2);       //  67 MB
  _Float16* Bu   = (_Float16*)carve((size_t)MROWS * 512 * 2);        // 134 MB
  _Float16* xs   = (_Float16*)carve((size_t)MROWS * 512 * 2);        // 134 MB
  float*    gout = (float*)   carve((size_t)MROWS * HDIM * 4);       // 134 MB
  _Float16* t1   = Bu;  // Bu dead after the scan; reuse for FFN activations.

  dim3 blk256(256);
  embed_mask<<<MROWS, blk256, 0, stream>>>(seqs, item_emb, pos_emb, x);

  for (int i = 0; i < NBLK; ++i) {
    prep_weights<<<512, blk256, 0, stream>>>(i, logAr, Aim, Bssm, Cssm, logstep,
                                             W1, W2, Abar, Wb, Wc, W1t, W2t);
    ln_rows<<<MROWS / 8, blk256, 0, stream>>>(x, ln_g + i * HDIM, ln_b + i * HDIM,
                                              hbuf, 1, 0);
    { dim3 g(MROWS / 256, 512 / 64);   // Bu = h @ Wb^T  -> f16
      gemm_f16<<<g, blk256, 64 * 256 * 2, stream>>>(hbuf, Wb, nullptr, Bu,
                                                    nullptr, 0, 512, 256); }
    s4_scan<<<B_SZ, blk256, 0, stream>>>(Bu, xs, Abar);
    { dim3 g(MROWS / 256, 256 / 64);   // y = xs @ Wc^T  -> f32
      gemm_f16<<<g, blk256, 64 * 512 * 2, stream>>>(xs, Wc, gout, nullptr,
                                                    nullptr, 0, 256, 512); }
    resid_s4_ln<<<MROWS / 8, blk256, 0, stream>>>(x, gout, hbuf, Dssm,
                                                  ln_g + i * HDIM,
                                                  ln_b + i * HDIM, hbuf, i);
    { dim3 g(MROWS / 256, 256 / 64);   // t1 = relu(h2 @ W1 + b1) -> f16
      gemm_f16<<<g, blk256, 64 * 256 * 2, stream>>>(hbuf, W1t, nullptr, t1,
                                                    b1 + i * HDIM, 1, 256, 256); }
    { dim3 g(MROWS / 256, 256 / 64);   // y2 = t1 @ W2 + b2 -> f32
      gemm_f16<<<g, blk256, 64 * 256 * 2, stream>>>(t1, W2t, gout, nullptr,
                                                    b2 + i * HDIM, 0, 256, 256); }
    resid_ffn_mask<<<MROWS, blk256, 0, stream>>>(x, gout, seqs);
  }

  // Final LN only on the last timestep of each batch, then candidate dots.
  ln_rows<<<B_SZ / 8, blk256, 0, stream>>>(x, lnfg, lnfb, xl, SEQ, SEQ - 1);
  logits_kernel<<<(B_SZ * NCAND) / 8, blk256, 0, stream>>>(xl, item_emb, idxs,
                                                           (float*)d_out);
}
